// SimpleMoE_893353198458
// MI455X (gfx1250) — matrix operand order repaired and checked by reading, not compiled, not run on hardware
//
#include <hip/hip_runtime.h>
#include <hip/hip_bf16.h>
#include <math.h>

// ---------------- problem constants (match reference) ----------------
#define NTOK   8192        // B*T
#define DDIM   1024
#define HDIM   4096
#define NEXP   8
#define TOPK   2
#define NSLOT  (NTOK*TOPK) // 16384

// ---------------- GEMM tiling ----------------
#define BM 128
#define BN 128
#define BKT 64            // K-tile (2 wmma k-steps of 32)
#define THREADS 256
#define AST 72            // padded LDS row stride (halves): 144B, 16B aligned, bank-spread
#define BST 72
#define MAX_MT (NSLOT/BM + NEXP)   // 136 worst-case M-tiles across all experts

typedef __attribute__((ext_vector_type(16))) __bf16 v16bf;
typedef __attribute__((ext_vector_type(8)))  float  v8f;
typedef int v4i_raw __attribute__((vector_size(16)));   // native int4 for async builtin

// ---- optional CDNA5 async memory->LDS path (ASYNCcnt-tracked) ----
#if defined(__AMDGCN__) && __has_builtin(__builtin_amdgcn_global_load_async_to_lds_b128)
#define ASYNC_A 1
#else
#define ASYNC_A 0
#endif

#if ASYNC_A
__device__ __forceinline__ void async_b128(const unsigned short* g, unsigned short* l) {
  __builtin_amdgcn_global_load_async_to_lds_b128(
      (__attribute__((address_space(1))) v4i_raw*)g,
      (__attribute__((address_space(3))) v4i_raw*)l, 0, 0);
}
# if __has_builtin(__builtin_amdgcn_s_wait_asynccnt)
#  define WAIT_ASYNC0() __builtin_amdgcn_s_wait_asynccnt(0)
# else
#  define WAIT_ASYNC0() asm volatile("s_wait_asynccnt 0" ::: "memory")
# endif
#else
#define WAIT_ASYNC0()
#endif

__device__ __forceinline__ unsigned short f2bf(float f) {
  unsigned u = __float_as_uint(f);
  u += 0x7FFFu + ((u >> 16) & 1u);           // round-to-nearest-even
  return (unsigned short)(u >> 16);
}

// ---------------- f32 -> bf16 bulk convert (vectorized) ----------------
__global__ void moe_cvt_bf16(const float* __restrict__ in,
                             unsigned short* __restrict__ out, int n4) {
  int i = blockIdx.x * blockDim.x + threadIdx.x;
  if (i >= n4) return;
  float4 v = ((const float4*)in)[i];
  ushort4 o;
  o.x = f2bf(v.x); o.y = f2bf(v.y); o.z = f2bf(v.z); o.w = f2bf(v.w);
  ((ushort4*)out)[i] = o;
}

// ---------------- router: logits, top-2, gates, per-expert counts ----------------
__global__ void moe_router(const float* __restrict__ x,
                           const float* __restrict__ rw,
                           const float* __restrict__ rb,
                           const float* __restrict__ ebias,
                           int* __restrict__ eidx, float* __restrict__ gout,
                           int* __restrict__ cnt) {
  int t = blockIdx.x * blockDim.x + threadIdx.x;
  if (t >= NTOK) return;
  const float4* xr = (const float4*)(x + (size_t)t * DDIM);
  float lg[NEXP];
  #pragma unroll
  for (int e = 0; e < NEXP; ++e) lg[e] = rb[e];
  for (int d4 = 0; d4 < DDIM / 4; ++d4) {
    float4 xv = xr[d4];
    #pragma unroll
    for (int e = 0; e < NEXP; ++e) {
      float4 wv = ((const float4*)(rw + (size_t)e * DDIM))[d4];
      lg[e] += xv.x * wv.x + xv.y * wv.y + xv.z * wv.z + xv.w * wv.w;
    }
  }
  float bb[NEXP];
  #pragma unroll
  for (int e = 0; e < NEXP; ++e) bb[e] = lg[e] + ebias[e];
  int e0 = 0;
  #pragma unroll
  for (int e = 1; e < NEXP; ++e) if (bb[e] > bb[e0]) e0 = e;   // ties -> lowest idx
  int e1 = -1;
  #pragma unroll
  for (int e = 0; e < NEXP; ++e)
    if (e != e0 && (e1 < 0 || bb[e] > bb[e1])) e1 = e;
  float r0 = lg[e0], r1 = lg[e1];
  float mx = fmaxf(r0, r1);
  float x0 = expf(r0 - mx), x1 = expf(r1 - mx);
  float inv = 1.0f / (x0 + x1);
  eidx[t * 2 + 0] = e0;  eidx[t * 2 + 1] = e1;
  gout[t * 2 + 0] = x0 * inv;  gout[t * 2 + 1] = x1 * inv;
  atomicAdd(&cnt[e0], 1);
  atomicAdd(&cnt[e1], 1);
}

// ---------------- tiny scan: expert offsets + counts output ----------------
__global__ void moe_scan(const int* __restrict__ cnt, int* __restrict__ offs,
                         int* __restrict__ cursor, int* __restrict__ counts_out) {
  if (threadIdx.x == 0 && blockIdx.x == 0) {
    int a = 0;
    for (int e = 0; e < NEXP; ++e) {
      offs[e] = a;
      counts_out[e] = cnt[e];
      cursor[e] = 0;
      a += cnt[e];
    }
    offs[NEXP] = a;
  }
}

// ---------------- scatter tokens into per-expert compacted slot lists ----------------
__global__ void moe_scatter(const int* __restrict__ eidx, const int* __restrict__ offs,
                            int* __restrict__ cursor, int* __restrict__ slot_tok,
                            int* __restrict__ slot_of) {
  int t = blockIdx.x * blockDim.x + threadIdx.x;
  if (t >= NTOK) return;
  #pragma unroll
  for (int k = 0; k < TOPK; ++k) {
    int e = eidx[t * 2 + k];
    int pos = atomicAdd(&cursor[e], 1);
    int s = offs[e] + pos;
    slot_tok[s] = t;
    slot_of[t * 2 + k] = s;
  }
}

// ---------------- grouped GEMM over experts (bf16 WMMA, f32 acc) ----------------
// MODE 1: A = xh[slot_tok[s]], W = w1h[e], K=D, cols=H, epilogue +b1,gelu -> bf16 hbuf
// MODE 2: A = hbuf[s],         W = w2h[e], K=H, cols=D, epilogue raw f32 -> ybuf
template <int MODE>
__global__ __launch_bounds__(THREADS)
void moe_ffn_gemm(const unsigned short* __restrict__ Abase,
                  const unsigned short* __restrict__ Wbase,
                  const float* __restrict__ bias,
                  const int* __restrict__ offs,
                  const int* __restrict__ slot_tok,
                  unsigned short* __restrict__ out_bf,
                  float* __restrict__ out_f32) {
  constexpr int KD  = (MODE == 1) ? DDIM : HDIM;   // reduction length
  constexpr int LDB = (MODE == 1) ? HDIM : DDIM;   // W row stride / out cols

  // ---- locate (expert, m-tile) for this block (uniform) ----
  int bx = blockIdx.x;
  int e = -1, mt = 0, a = 0;
  #pragma unroll
  for (int i = 0; i < NEXP; ++i) {
    int c  = offs[i + 1] - offs[i];
    int tl = (c + BM - 1) / BM;
    if (e < 0 && bx < a + tl) { e = i; mt = bx - a; }
    a += tl;
  }
  if (e < 0) return;                       // uniform exit: EXEC stays full below
  const int sBeg  = offs[e] + mt * BM;
  const int sEnd  = offs[e + 1];
  const int nBase = blockIdx.y * BN;

  __shared__ unsigned short As[2][BM * AST];  // row-major [m][k], padded
  __shared__ unsigned short Bs[2][BN * BST];  // transposed [n][k], padded
  __shared__ int sRow[BM];

  const int tid = threadIdx.x;
  for (int r = tid; r < BM; r += THREADS) {
    int s  = sBeg + r;
    int sc = (s < sEnd) ? s : (sEnd - 1);  // clamp: keeps loads in-bounds
    sRow[r] = (MODE == 1) ? slot_tok[sc] : sc;
  }
  __syncthreads();

  const int wave  = tid >> 5;
  const int lane  = tid & 31;
  const int wm    = wave & 1;              // 2 wave-rows * 64
  const int wn    = wave >> 1;             // 4 wave-cols * 32
  const int lhalf = lane >> 4;             // 0/1
  const int l15   = lane & 15;

  v8f acc[4][2];
  #pragma unroll
  for (int i = 0; i < 4; ++i)
    #pragma unroll
    for (int j = 0; j < 2; ++j)
      #pragma unroll
      for (int q = 0; q < 8; ++q) acc[i][j][q] = 0.0f;

  // ---- per-thread load geometry ----
  // A: 2 threads/row, 32 halves (64B = 4xb128) each
  const int arow  = tid >> 1;
  const int acolh = (tid & 1) * 32;
  const size_t aoff = (size_t)sRow[arow] * KD + acolh;
  unsigned short* aLds0 = &As[0][arow * AST + acolh];
  unsigned short* aLds1 = &As[1][arow * AST + acolh];
  // B: wave kq owns k-rows [kq*8, kq*8+8), lane ng owns 4 cols; transpose in regs
  const int kq = tid >> 5;
  const int ng = tid & 31;
  const unsigned short* Wexp = Wbase + (size_t)e * KD * LDB;

  ushort4 bw[8];
  auto loadB = [&](int k0) {
    #pragma unroll
    for (int i = 0; i < 8; ++i)
      bw[i] = *(const ushort4*)(Wexp + (size_t)(k0 + kq * 8 + i) * LDB + nBase + ng * 4);
  };
  auto storeB = [&](int buf) {
    #pragma unroll
    for (int c = 0; c < 4; ++c) {
      union { unsigned short u[8]; uint4 q; } t;
      #pragma unroll
      for (int i = 0; i < 8; ++i) t.u[i] = ((const unsigned short*)&bw[i])[c];
      *(uint4*)&Bs[buf][(ng * 4 + c) * BST + kq * 8] = t.q;   // 16B aligned b128
    }
  };
#if ASYNC_A
  auto loadA_async = [&](int k0, int buf) {
    const unsigned short* g = Abase + aoff + k0;
    unsigned short* l = buf ? aLds1 : aLds0;
    #pragma unroll
    for (int i = 0; i < 4; ++i) async_b128(g + i * 8, l + i * 8);
  };
#else
  uint4 aw[4];
  auto loadA = [&](int k0) {
    const uint4* g = (const uint4*)(Abase + aoff + k0);
    #pragma unroll
    for (int i = 0; i < 4; ++i) aw[i] = g[i];
  };
  auto storeA = [&](int buf) {
    uint4* l = (uint4*)(buf ? aLds1 : aLds0);
    #pragma unroll
    for (int i = 0; i < 4; ++i) l[i] = aw[i];
  };
#endif

  constexpr int NIT = KD / BKT;

  // ---- prologue: stage tile 0 into buffer 0 ----
  loadB(0);
#if ASYNC_A
  loadA_async(0, 0);
#else
  loadA(0);
  storeA(0);
#endif
  storeB(0);

  for (int it = 0; it < NIT; ++it) {
    const int cur  = it & 1;
    const int nxt  = cur ^ 1;
    const bool more = (it + 1 < NIT);

    if (more) loadB((it + 1) * BKT);       // global->regs, issue early
#if !ASYNC_A
    if (more) loadA((it + 1) * BKT);
#endif
    WAIT_ASYNC0();                          // this wave's async A for tile `it` landed
    __syncthreads();                        // tile `it` fully visible in LDS
#if ASYNC_A
    if (more) loadA_async((it + 1) * BKT, nxt);  // overlap with wmma below
#endif

    // ---- compute: 2 k-steps of 32, 16 wmma ----
    #pragma unroll
    for (int kk = 0; kk < 2; ++kk) {
      union FA { v16bf v; uint4 q[2]; } fa[4];
      #pragma unroll
      for (int i = 0; i < 4; ++i) {
        const unsigned short* p =
            &As[cur][(wm * 64 + i * 16 + l15) * AST + kk * 32 + lhalf * 8];
        fa[i].q[0] = *(const uint4*)(p);        // K = ksel*8 .. +8
        fa[i].q[1] = *(const uint4*)(p + 16);   // K = 16+ksel*8 .. +8
      }
      union FB { v16bf v; uint4 q[2]; } fb[2];
      #pragma unroll
      for (int j = 0; j < 2; ++j) {
        const unsigned short* p =
            &Bs[cur][(wn * 32 + j * 16 + l15) * BST + kk * 32 + lhalf * 8];
        fb[j].q[0] = *(const uint4*)(p);        // K = ksel*8 .. +8
        fb[j].q[1] = *(const uint4*)(p + 16);
      }
      #pragma unroll
      for (int i = 0; i < 4; ++i)
        #pragma unroll
        for (int j = 0; j < 2; ++j)
          acc[i][j] = __builtin_amdgcn_wmma_f32_16x16x32_bf16(
              false, fa[i].v, false, fb[j].v, (short)0, acc[i][j], false, false);
    }

    if (more) {
      storeB(nxt);                          // regs->LDS for tile it+1
#if !ASYNC_A
      storeA(nxt);
#endif
    }
    __syncthreads();                        // readers of `cur` done -> reusable
  }

  // ---- epilogue (guarded stores only; C/D layout: m = r + 8*lhalf, n = l15) ----
  #pragma unroll
  for (int i = 0; i < 4; ++i) {
    #pragma unroll
    for (int j = 0; j < 2; ++j) {
      const int ncol = nBase + wn * 32 + j * 16 + l15;
      #pragma unroll
      for (int r = 0; r < 8; ++r) {
        const int m = wm * 64 + i * 16 + lhalf * 8 + r;
        const int s = sBeg + m;
        if (s < sEnd) {
          float c = acc[i][j][r];
          if (MODE == 1) {
            float v = c + bias[(size_t)e * HDIM + ncol];
            v = 0.5f * v * (1.0f + erff(v * 0.70710678118654752f));  // exact gelu
            out_bf[(size_t)s * HDIM + ncol] = f2bf(v);
          } else {
            out_f32[(size_t)s * DDIM + ncol] = c;
          }
        }
      }
    }
  }
}

// ---------------- gate/bias combine: y[t] = sum_k g_k*(ybuf[slot]+b2[e_k]) ----------------
__global__ void moe_combine(const float* __restrict__ ybuf, const float* __restrict__ b2,
                            const int* __restrict__ eidx, const float* __restrict__ gout,
                            const int* __restrict__ slot_of, float* __restrict__ y) {
  int i = blockIdx.x * blockDim.x + threadIdx.x;
  if (i >= NTOK * (DDIM / 4)) return;
  int t  = i / (DDIM / 4);
  int c4 = i % (DDIM / 4);
  int e0 = eidx[t * 2], e1 = eidx[t * 2 + 1];
  float g0 = gout[t * 2], g1 = gout[t * 2 + 1];
  int s0 = slot_of[t * 2], s1 = slot_of[t * 2 + 1];
  float4 c0 = ((const float4*)(ybuf + (size_t)s0 * DDIM))[c4];
  float4 c1 = ((const float4*)(ybuf + (size_t)s1 * DDIM))[c4];
  float4 b0 = ((const float4*)(b2 + (size_t)e0 * DDIM))[c4];
  float4 b1v = ((const float4*)(b2 + (size_t)e1 * DDIM))[c4];
  float4 r;
  r.x = g0 * (c0.x + b0.x) + g1 * (c1.x + b1v.x);
  r.y = g0 * (c0.y + b0.y) + g1 * (c1.y + b1v.y);
  r.z = g0 * (c0.z + b0.z) + g1 * (c1.z + b1v.z);
  r.w = g0 * (c0.w + b0.w) + g1 * (c1.w + b1v.w);
  ((float4*)(y + (size_t)t * DDIM))[c4] = r;
}

// ---------------- host entry ----------------
extern "C" void kernel_launch(void* const* d_in, const int* in_sizes, int n_in,
                              void* d_out, int out_size, void* d_ws, size_t ws_size,
                              hipStream_t stream) {
  const float* x  = (const float*)d_in[0];
  const float* rw = (const float*)d_in[1];
  const float* rb = (const float*)d_in[2];
  const float* eb = (const float*)d_in[3];
  const float* w1 = (const float*)d_in[4];
  const float* b1 = (const float*)d_in[5];
  const float* w2 = (const float*)d_in[6];
  const float* b2 = (const float*)d_in[7];
  float* y = (float*)d_out;
  int* counts_out = (int*)((float*)d_out + (size_t)NTOK * DDIM);

  // workspace carve-up (256B aligned)
  char* ws = (char*)d_ws;
  size_t off = 0;
  auto carve = [&](size_t bytes) -> char* {
    off = (off + 255) & ~(size_t)255;
    char* p = ws + off;
    off += bytes;
    return p;
  };
  unsigned short* xh   = (unsigned short*)carve((size_t)NTOK * DDIM * 2);
  unsigned short* w1h  = (unsigned short*)carve((size_t)NEXP * DDIM * HDIM * 2);
  unsigned short* w2h  = (unsigned short*)carve((size_t)NEXP * HDIM * DDIM * 2);
  unsigned short* hbuf = (unsigned short*)carve((size_t)NSLOT * HDIM * 2);
  float*          ybuf = (float*)carve((size_t)NSLOT * DDIM * 4);
  int* eidx     = (int*)carve((size_t)NTOK * 2 * 4);
  float* gout   = (float*)carve((size_t)NTOK * 2 * 4);
  int* slot_tok = (int*)carve((size_t)NSLOT * 4);
  int* slot_of  = (int*)carve((size_t)NTOK * 2 * 4);
  int* cnt      = (int*)carve(NEXP * 4);
  int* offs     = (int*)carve((NEXP + 1) * 4);
  int* cursor   = (int*)carve(NEXP * 4);

  (void)hipMemsetAsync(cnt, 0, NEXP * sizeof(int), stream);

  // bf16 conversions
  {
    int n4 = NTOK * DDIM / 4;
    moe_cvt_bf16<<<(n4 + 255) / 256, 256, 0, stream>>>(x, xh, n4);
    n4 = NEXP * DDIM * HDIM / 4;
    moe_cvt_bf16<<<(n4 + 255) / 256, 256, 0, stream>>>(w1, w1h, n4);
    moe_cvt_bf16<<<(n4 + 255) / 256, 256, 0, stream>>>(w2, w2h, n4);
  }

  moe_router<<<NTOK / 256, 256, 0, stream>>>(x, rw, rb, eb, eidx, gout, cnt);
  moe_scan<<<1, 32, 0, stream>>>(cnt, offs, cursor, counts_out);
  moe_scatter<<<NTOK / 256, 256, 0, stream>>>(eidx, offs, cursor, slot_tok, slot_of);

  dim3 g1(MAX_MT, HDIM / BN);   // 136 x 32
  moe_ffn_gemm<1><<<g1, THREADS, 0, stream>>>(xh, w1h, b1, offs, slot_tok, hbuf, nullptr);
  dim3 g2(MAX_MT, DDIM / BN);   // 136 x 8
  moe_ffn_gemm<2><<<g2, THREADS, 0, stream>>>(hbuf, w2h, nullptr, offs, slot_tok, nullptr, ybuf);

  int nc = NTOK * (DDIM / 4);
  moe_combine<<<(nc + 255) / 256, 256, 0, stream>>>(ybuf, b2, eidx, gout, slot_of, y);
}
